// TransformerLikeGRUBlock_86603720556887
// MI455X (gfx1250) — matrix operand order repaired and checked by reading, not compiled, not run on hardware
//
#include <hip/hip_runtime.h>

// ---------------------------------------------------------------------------
// Types for CDNA5 WMMA (wave32): v_wmma_f32_16x16x32_bf16
// ---------------------------------------------------------------------------
typedef __attribute__((ext_vector_type(16))) __bf16 v16bf;
typedef __attribute__((ext_vector_type(8)))  float  v8f;

union FragB {
  v16bf v;
  uint4 q[2];
};

__device__ __forceinline__ unsigned short f2bf(float f) {
  unsigned u = __float_as_uint(f);
  u += 0x7FFFu + ((u >> 16) & 1u);      // round-to-nearest-even
  return (unsigned short)(u >> 16);
}

// Async global->LDS copy of 32 contiguous bytes (2 x b128), tracked by ASYNCcnt.
// INST_OFFSET applies to both the LDS and global address (ISA 08_async_tensor).
__device__ __forceinline__ void async_copy_b128x2(unsigned lds_off,
                                                  const void* gaddr) {
  asm volatile(
      "global_load_async_to_lds_b128 %0, %1, off\n\t"
      "global_load_async_to_lds_b128 %0, %1, off offset:16"
      :: "v"(lds_off), "v"(gaddr) : "memory");
}
__device__ __forceinline__ void wait_async0() {
  asm volatile("s_wait_asynccnt 0x0" ::: "memory");
}

// ---------------------------------------------------------------------------
// fp32 -> bf16 transposed conversion (weights): in (K,N) f32 -> out (N,K) bf16
// LDS-tiled 32x32 so both global streams stay coalesced.
// ---------------------------------------------------------------------------
__global__ __launch_bounds__(256) void cvt_t_bf16_kernel(
    const float* __restrict__ in, unsigned short* __restrict__ out, int K,
    int N) {
  __shared__ float tile[32][33];
  const int kb = blockIdx.y * 32, nb = blockIdx.x * 32;
  const int tx = threadIdx.x & 31, ty = threadIdx.x >> 5;  // 32 x 8
#pragma unroll
  for (int i = 0; i < 32; i += 8)
    tile[ty + i][tx] = in[(size_t)(kb + ty + i) * N + nb + tx];
  __syncthreads();
#pragma unroll
  for (int i = 0; i < 32; i += 8)
    out[(size_t)(nb + ty + i) * K + kb + tx] = f2bf(tile[tx][ty + i]);
}

// ---------------------------------------------------------------------------
// LayerNorm over D=1024, output bf16
// ---------------------------------------------------------------------------
__global__ __launch_bounds__(256) void ln_bf16_kernel(
    const float* __restrict__ x, const float* __restrict__ g,
    const float* __restrict__ b, unsigned short* __restrict__ out) {
  __shared__ float red[512];
  const int row = blockIdx.x;
  const float* xr = x + (size_t)row * 1024;
  float v[4];
  float s = 0.f, s2 = 0.f;
#pragma unroll
  for (int i = 0; i < 4; i++) {
    v[i] = xr[threadIdx.x + i * 256];
    s += v[i];
    s2 += v[i] * v[i];
  }
  red[threadIdx.x] = s;
  red[256 + threadIdx.x] = s2;
  __syncthreads();
  for (int st = 128; st > 0; st >>= 1) {
    if ((int)threadIdx.x < st) {
      red[threadIdx.x] += red[threadIdx.x + st];
      red[256 + threadIdx.x] += red[256 + threadIdx.x + st];
    }
    __syncthreads();
  }
  const float mean = red[0] * (1.f / 1024.f);
  const float var = red[256] * (1.f / 1024.f) - mean * mean;
  const float rstd = rsqrtf(var + 1e-5f);
#pragma unroll
  for (int i = 0; i < 4; i++) {
    int c = threadIdx.x + i * 256;
    out[(size_t)row * 1024 + c] = f2bf((v[i] - mean) * rstd * g[c] + b[c]);
  }
}

// ---------------------------------------------------------------------------
// Tiled bf16 WMMA GEMM: C(MxN) = A(MxK) @ BT(NxK)^T + epilogue
//  - BT is the pre-transposed weight (N-major), so A and B tiles are both
//    contiguous 32B/thread copies, staged via async global->LDS (ASYNCcnt),
//    double-buffered so each copy overlaps one K-step of WMMA compute.
// Block = 256 threads (8 waves). Tile 128x128, BK=32.
// Wave (mw 0..3, nw 0..1) computes 32 rows x 64 cols = 2x4 WMMA tiles.
// EPI: 0 = +bias -> bf16 ; 1 = +bias, SiLU -> bf16 ; 2 = +bias, +residual -> f32
// ---------------------------------------------------------------------------
template <int EPI>
__global__ __launch_bounds__(256) void gemm_bf16_kernel(
    const unsigned short* __restrict__ A, const unsigned short* __restrict__ BT,
    const float* __restrict__ bias, const float* __restrict__ res,
    void* __restrict__ outv, int M, int Nc, int K) {
  __shared__ unsigned short As[2][128 * 40];  // 40 = 32 + 8 pad (80B rows)
  __shared__ unsigned short Bs[2][128 * 40];  // Bs[n][k]
  const int tid = threadIdx.x;
  const int lane = tid & 31, wave = tid >> 5;
  const int mw = wave & 3, nw = wave >> 2;
  const int lrow = lane & 15, khalf = lane >> 4;
  const long bm = (long)blockIdx.y * 128;
  const long bn = (long)blockIdx.x * 128;

  v8f acc[2][4] = {};

  // staging coords: 128 rows x 32 k, 16 bf16 (32B) per thread
  const int srow = tid >> 1, kseg = (tid & 1) << 4;
  const unsigned short* Ag = A + (size_t)(bm + srow) * K + kseg;
  const unsigned short* Bg = BT + (size_t)(bn + srow) * K + kseg;
  const unsigned ldsA[2] = {(unsigned)(size_t)&As[0][srow * 40 + kseg],
                            (unsigned)(size_t)&As[1][srow * 40 + kseg]};
  const unsigned ldsB[2] = {(unsigned)(size_t)&Bs[0][srow * 40 + kseg],
                            (unsigned)(size_t)&Bs[1][srow * 40 + kseg]};

  // prologue: stage tile 0
  async_copy_b128x2(ldsA[0], Ag);
  async_copy_b128x2(ldsB[0], Bg);
  Ag += 32;
  Bg += 32;
  wait_async0();
  __syncthreads();

  const int nTiles = K >> 5;
  for (int t = 0; t < nTiles; ++t) {
    const int cur = t & 1;
    if (t + 1 < nTiles) {  // overlap next tile's copy with this tile's math
      async_copy_b128x2(ldsA[cur ^ 1], Ag);
      async_copy_b128x2(ldsB[cur ^ 1], Bg);
      Ag += 32;
      Bg += 32;
    }
    const unsigned short* Asc = &As[cur][0];
    const unsigned short* Bsc = &Bs[cur][0];

    FragB afr[2];
#pragma unroll
    for (int i = 0; i < 2; i++) {
      const int m = mw * 32 + i * 16 + lrow;
      afr[i].q[0] = *(const uint4*)(&Asc[m * 40 + khalf * 8]);
      afr[i].q[1] = *(const uint4*)(&Asc[m * 40 + 16 + khalf * 8]);
    }
#pragma unroll
    for (int j = 0; j < 4; j++) {
      FragB bfr;
      const int n = nw * 64 + j * 16 + lrow;
      bfr.q[0] = *(const uint4*)(&Bsc[n * 40 + khalf * 8]);
      bfr.q[1] = *(const uint4*)(&Bsc[n * 40 + 16 + khalf * 8]);
#pragma unroll
      for (int i = 0; i < 2; i++) {
        acc[i][j] = __builtin_amdgcn_wmma_f32_16x16x32_bf16(
            false, afr[i].v, false, bfr.v, (short)0, acc[i][j], false, false);
      }
    }
    if (t + 1 < nTiles) {
      wait_async0();
      __syncthreads();
    }
  }

  // Epilogue: C/D layout -> col = 16*tile + lane%16, row = 8*(lane/16) + r
#pragma unroll
  for (int j = 0; j < 4; j++) {
    const long col = bn + nw * 64 + j * 16 + lrow;
    const float bv = bias[col];
#pragma unroll
    for (int i = 0; i < 2; i++) {
#pragma unroll
      for (int r = 0; r < 8; r++) {
        const long row = bm + mw * 32 + i * 16 + khalf * 8 + r;
        float vv = acc[i][j][r] + bv;
        if (EPI == 1) vv = vv / (1.f + __expf(-vv));  // SiLU
        if (EPI == 2) {
          vv += res[(size_t)row * Nc + col];
          ((float*)outv)[(size_t)row * Nc + col] = vv;
        } else {
          ((unsigned short*)outv)[(size_t)row * Nc + col] = f2bf(vv);
        }
      }
    }
  }
}

// ---------------------------------------------------------------------------
// Per-head gate kernel: for each (n,h) slab of pi (L x 64, bf16), compute
// k = pi@wz + bz, th = pi@wh + bh via WMMA, then fused log-space gate math:
//   logc  = -softplus(k)
//   lzth  = -softplus(-k) + log_g(th)
// wzT/whT arrive pre-transposed (N,K) bf16. Output layout (NH, L, 64) f32.
// ---------------------------------------------------------------------------
__global__ __launch_bounds__(256) void head_gates_kernel(
    const unsigned short* __restrict__ pi, const unsigned short* __restrict__ wzT,
    const float* __restrict__ bz, const unsigned short* __restrict__ whT,
    const float* __restrict__ bh, float* __restrict__ logc,
    float* __restrict__ lzth) {
  __shared__ unsigned short WzT[64 * 72];  // WzT[n][k], 144B rows
  __shared__ unsigned short WhT[64 * 72];
  const int tid = threadIdx.x;
  {
    const int n_ = tid >> 2, ks_ = (tid & 3) << 4;  // 64 rows x 4 chunks
    *(uint4*)&WzT[n_ * 72 + ks_] = *(const uint4*)(wzT + n_ * 64 + ks_);
    *(uint4*)&WzT[n_ * 72 + ks_ + 8] = *(const uint4*)(wzT + n_ * 64 + ks_ + 8);
    *(uint4*)&WhT[n_ * 72 + ks_] = *(const uint4*)(whT + n_ * 64 + ks_);
    *(uint4*)&WhT[n_ * 72 + ks_ + 8] = *(const uint4*)(whT + n_ * 64 + ks_ + 8);
  }
  __syncthreads();

  const int lane = tid & 31, wave = tid >> 5;
  const int lrow = lane & 15, khalf = lane >> 4;
  const int nh = blockIdx.y;
  const int nb = nh >> 4, hh = nh & 15;
  const long t0 = (long)blockIdx.x * 128 + wave * 16;

  const unsigned short* arow =
      pi + ((size_t)nb * 4096 + t0 + lrow) * 1024 + hh * 64;

  v8f acck[4] = {};
  v8f acch[4] = {};
#pragma unroll
  for (int ks = 0; ks < 2; ks++) {
    const int k0 = ks * 32;
    FragB a;
    a.q[0] = *(const uint4*)(arow + k0 + khalf * 8);
    a.q[1] = *(const uint4*)(arow + k0 + 16 + khalf * 8);
#pragma unroll
    for (int j = 0; j < 4; j++) {
      const int n = j * 16 + lrow;
      FragB bz_, bh_;
      bz_.q[0] = *(const uint4*)(&WzT[n * 72 + k0 + khalf * 8]);
      bz_.q[1] = *(const uint4*)(&WzT[n * 72 + k0 + 16 + khalf * 8]);
      bh_.q[0] = *(const uint4*)(&WhT[n * 72 + k0 + khalf * 8]);
      bh_.q[1] = *(const uint4*)(&WhT[n * 72 + k0 + 16 + khalf * 8]);
      acck[j] = __builtin_amdgcn_wmma_f32_16x16x32_bf16(
          false, a.v, false, bz_.v, (short)0, acck[j], false, false);
      acch[j] = __builtin_amdgcn_wmma_f32_16x16x32_bf16(
          false, a.v, false, bh_.v, (short)0, acch[j], false, false);
    }
  }

#pragma unroll
  for (int j = 0; j < 4; j++) {
    const int col = j * 16 + lrow;
    const float bzv = bz[col], bhv = bh[col];
#pragma unroll
    for (int r = 0; r < 8; r++) {
      const long t = t0 + khalf * 8 + r;
      const float kv = acck[j][r] + bzv;
      const float thv = acch[j][r] + bhv;
      const float log_z = -log1pf(expf(-kv));   // -softplus(-k)
      const float lcv = -log1pf(expf(kv));      // -softplus(k)
      const float lth =
          (thv >= 0.f) ? logf(thv + 0.5f) : -log1pf(expf(-thv));  // log_g
      const size_t o = ((size_t)nh * 4096 + t) * 64 + col;
      logc[o] = lcv;
      lzth[o] = log_z + lth;
    }
  }
}

// ---------------------------------------------------------------------------
// Sequential minGRU recurrence (real-space equivalent of the log-scan):
//   h_t = exp(logc_t) * h_{t-1} + exp(lzth_t),  h_init = g(h0)
// 4096 independent channels. Writes rnn_states (f32, output #2) and a bf16
// copy for the w_out GEMM.
// ---------------------------------------------------------------------------
__global__ __launch_bounds__(256) void scan_kernel(
    const float* __restrict__ h0, const float* __restrict__ logc,
    const float* __restrict__ lzth, float* __restrict__ outh,
    unsigned short* __restrict__ hbf) {
  const int gid = blockIdx.x * 256 + threadIdx.x;  // 0..4095
  const int seq = gid >> 6, d = gid & 63;
  const int nb = seq >> 4, hh = seq & 15;
  const float x0 = h0[(size_t)nb * 1024 + hh * 64 + d];
  float hv = (x0 >= 0.f) ? (x0 + 0.5f) : (1.f / (1.f + expf(-x0)));  // g(h0)
  const float* lc = logc + (size_t)seq * 4096 * 64 + d;
  const float* lz = lzth + (size_t)seq * 4096 * 64 + d;
  float* op = outh + (size_t)nb * 4096 * 1024 + hh * 64 + d;
  unsigned short* hp = hbf + (size_t)nb * 4096 * 1024 + hh * 64 + d;
  for (int t = 0; t < 4096; t++) {
    hv = expf(lc[(size_t)t * 64]) * hv + expf(lz[(size_t)t * 64]);
    op[(size_t)t * 1024] = hv;
    hp[(size_t)t * 1024] = f2bf(hv);
  }
}

// ---------------------------------------------------------------------------
// Host orchestration
// ---------------------------------------------------------------------------
extern "C" void kernel_launch(void* const* d_in, const int* in_sizes, int n_in,
                              void* d_out, int out_size, void* d_ws,
                              size_t ws_size, hipStream_t stream) {
  (void)in_sizes; (void)n_in; (void)out_size; (void)ws_size;
  const float* x     = (const float*)d_in[0];
  const float* h0    = (const float*)d_in[1];
  const float* w_in  = (const float*)d_in[2];
  const float* b_in  = (const float*)d_in[3];
  const float* wz    = (const float*)d_in[4];
  const float* bz    = (const float*)d_in[5];
  const float* wh    = (const float*)d_in[6];
  const float* bh    = (const float*)d_in[7];
  const float* w_out = (const float*)d_in[8];
  const float* b_out = (const float*)d_in[9];
  const float* ln1g  = (const float*)d_in[10];
  const float* ln1b  = (const float*)d_in[11];
  const float* ln2g  = (const float*)d_in[12];
  const float* ln2b  = (const float*)d_in[13];
  const float* mw1   = (const float*)d_in[14];
  const float* mb1   = (const float*)d_in[15];
  const float* mw2   = (const float*)d_in[16];
  const float* mb2   = (const float*)d_in[17];
  const float* mw3   = (const float*)d_in[18];
  const float* mb3   = (const float*)d_in[19];

  constexpr size_t MR = 16384;   // N*L rows
  constexpr size_t D = 1024, MLPD = 4096, L = 4096, NH = 64;

  char* ws = (char*)d_ws;
  size_t off = 0;
  auto alloc = [&](size_t bytes) {
    size_t o = off;
    off += (bytes + 255) & ~(size_t)255;
    return o;
  };
  // bf16 transposed weights (persistent)
  const size_t o_win = alloc(D * D * 2);
  const size_t o_wz  = alloc(64 * 64 * 2);
  const size_t o_wh  = alloc(64 * 64 * 2);
  const size_t o_wout = alloc(D * D * 2);
  const size_t o_mw1 = alloc(D * MLPD * 2);
  const size_t o_mw2 = alloc(MLPD * MLPD * 2);
  const size_t o_mw3 = alloc(MLPD * D * 2);
  // activations (with reuse)
  const size_t o_A = alloc(MR * D * 2);     // xln bf16  -> later h bf16
  const size_t o_B = alloc(MR * D * 2);     // pi bf16   -> later x2ln bf16
  const size_t o_C = alloc(MR * D * 4);     // logc f32  -> later x2 f32
  const size_t o_Dd = alloc(MR * D * 4);    // lzth f32
  const size_t o_E = alloc(MR * MLPD * 2);  // y1 bf16
  const size_t o_F = alloc(MR * MLPD * 2);  // y2 bf16

  auto U16 = [&](size_t o) { return (unsigned short*)(ws + o); };
  auto F32 = [&](size_t o) { return (float*)(ws + o); };

  // weights: fp32 (K,N) -> bf16 transposed (N,K)
  auto cvt_t = [&](const float* src, size_t dsto, size_t K, size_t N) {
    cvt_t_bf16_kernel<<<dim3((unsigned)(N / 32), (unsigned)(K / 32)), 256, 0,
                        stream>>>(src, U16(dsto), (int)K, (int)N);
  };
  cvt_t(w_in, o_win, D, D);
  cvt_t(wz, o_wz, 64, 64);
  cvt_t(wh, o_wh, 64, 64);
  cvt_t(w_out, o_wout, D, D);
  cvt_t(mw1, o_mw1, D, MLPD);
  cvt_t(mw2, o_mw2, MLPD, MLPD);
  cvt_t(mw3, o_mw3, MLPD, D);

  float* out1 = (float*)d_out;            // x + y
  float* out2 = out1 + MR * D;            // rnn_states

  // LN1 -> bf16
  ln_bf16_kernel<<<(unsigned)MR, 256, 0, stream>>>(x, ln1g, ln1b, U16(o_A));
  // pi = xln @ w_in + b_in  (bf16 out)
  gemm_bf16_kernel<0><<<dim3((unsigned)(D / 128), (unsigned)(MR / 128)), 256, 0,
                        stream>>>(U16(o_A), U16(o_win), b_in, nullptr,
                                  U16(o_B), (int)MR, (int)D, (int)D);
  // gate pre-activations + log-space gate math
  head_gates_kernel<<<dim3((unsigned)(L / 128), (unsigned)NH), 256, 0,
                      stream>>>(U16(o_B), U16(o_wz), bz, U16(o_wh), bh,
                                F32(o_C), F32(o_Dd));
  // sequential recurrence -> rnn_states (f32) + bf16 copy
  scan_kernel<<<16, 256, 0, stream>>>(h0, F32(o_C), F32(o_Dd), out2, U16(o_A));
  // x2 = x + h @ w_out + b_out  (f32 out)
  gemm_bf16_kernel<2><<<dim3((unsigned)(D / 128), (unsigned)(MR / 128)), 256, 0,
                        stream>>>(U16(o_A), U16(o_wout), b_out, x, F32(o_C),
                                  (int)MR, (int)D, (int)D);
  // LN2 -> bf16
  ln_bf16_kernel<<<(unsigned)MR, 256, 0, stream>>>(F32(o_C), ln2g, ln2b,
                                                   U16(o_B));
  // y1 = silu(x2ln @ mw1 + mb1)
  gemm_bf16_kernel<1><<<dim3((unsigned)(MLPD / 128), (unsigned)(MR / 128)), 256,
                        0, stream>>>(U16(o_B), U16(o_mw1), mb1, nullptr,
                                     U16(o_E), (int)MR, (int)MLPD, (int)D);
  // y2 = silu(y1 @ mw2 + mb2)
  gemm_bf16_kernel<1><<<dim3((unsigned)(MLPD / 128), (unsigned)(MR / 128)), 256,
                        0, stream>>>(U16(o_E), U16(o_mw2), mb2, nullptr,
                                     U16(o_F), (int)MR, (int)MLPD, (int)MLPD);
  // out1 = x2 + y2 @ mw3 + mb3
  gemm_bf16_kernel<2><<<dim3((unsigned)(D / 128), (unsigned)(MR / 128)), 256, 0,
                        stream>>>(U16(o_F), U16(o_mw3), mb3, F32(o_C), out1,
                                  (int)MR, (int)D, (int)MLPD);
}
